// DenoisedSasrec_12257836663480
// MI455X (gfx1250) — matrix operand order repaired and checked by reading, not compiled, not run on hardware
//
#include <hip/hip_runtime.h>
#include <hip/hip_bf16.h>

// ---------------------------------------------------------------------------
// DenoisedSasrec on MI455X (gfx1250): six 1024-aligned GEMMs via
// v_wmma_f32_16x16x32_bf16 (wave32), bf16 storage / f32 accumulation.
// K-loop double-buffered with gfx1250 async global->LDS DMA
// (global_load_async_to_lds_b128 / s_wait_asynccnt).
// ---------------------------------------------------------------------------

typedef __attribute__((ext_vector_type(16))) __bf16 v16bf;
typedef __attribute__((ext_vector_type(8)))  __bf16 v8bf;
typedef __attribute__((ext_vector_type(4)))  __bf16 v4bf;
typedef __attribute__((ext_vector_type(8)))  float  v8f;

#define LSEQ 1024

// Tiling: 128x256x32 block tile, 256 threads = 8 waves (2x4); wave tile 64x64
// = 4x4 WMMA tiles -> 16 v_wmma per K-step per wave vs 16 ds_load_b128.
constexpr int BM = 128;
constexpr int BN = 256;
constexpr int BK = 32;
constexpr int LDSROW = 40;   // 32 bf16 + 8 pad -> 80B row stride (16B aligned, bank-friendly)
constexpr int NROWS = BM + BN; // 384 rows per buffer: A rows [0,128), B rows [128,384)

enum Epi { EPI_SILU = 0, EPI_SCALE = 1, EPI_ATTN = 2, EPI_SILU_T = 3, EPI_OUT = 4 };

__device__ __forceinline__ float silu_f(float x) {
    return x * (1.0f / (1.0f + __expf(-x)));
}

// gfx1250 async DMA: global -> LDS, 16B, tracked by ASYNCcnt (bypasses VGPRs)
__device__ __forceinline__ void async_ld_b128(unsigned int lds_off, const void* g) {
    asm volatile("global_load_async_to_lds_b128 %0, %1, off"
                 :: "v"(lds_off), "v"((unsigned long long)(uintptr_t)g)
                 : "memory");
}
__device__ __forceinline__ void wait_async0() {
    asm volatile("s_wait_asynccnt 0x0" ::: "memory");
}

// A fragment (16x32 bf16): lane half=lane>>4, row=lane&15.
// elements [0..7]  = K half*8 .. half*8+7 ; [8..15] = K 16+half*8 ..
__device__ __forceinline__ v16bf load_frag_a(const __bf16* base, int row, int half) {
    union { v16bf v; v8bf h[2]; } u;
    const __bf16* p = base + row * LDSROW;
    u.h[0] = *(const v8bf*)(p + half * 8);
    u.h[1] = *(const v8bf*)(p + 16 + half * 8);
    return u.v;
}

// B fragment (32x16 bf16): lane half=lane>>4, col=lane&15.
// elements [0..7]  = K half*8 .. half*8+7 ; [8..15] = K 16+half*8 ..
__device__ __forceinline__ v16bf load_frag_b(const __bf16* base, int col, int half) {
    union { v16bf v; v8bf h[2]; } u;
    const __bf16* p = base + col * LDSROW;
    u.h[0] = *(const v8bf*)(p + half * 8);
    u.h[1] = *(const v8bf*)(p + 16 + half * 8);
    return u.v;
}

// Stream one 128x32 (A) + 256x32 (B) tile into an LDS buffer: 6 async b128/thread.
__device__ __forceinline__ void issue_tile_async(
    const __bf16* __restrict__ A, const __bf16* __restrict__ B,
    int bm, int bn, int K, int k0, const __bf16* ldsbuf, int t)
{
#pragma unroll
    for (int p = 0; p < 6; p++) {
        const int c = t + p * 256;          // 0..1535
        const int row = c >> 2;             // 0..383 (branch below is uniform per p)
        const int col = (c & 3) * 8;        // bf16 offset in row
        const __bf16* g = (row < BM)
            ? &A[(long long)(bm + row) * K + k0 + col]
            : &B[(long long)(bn + (row - BM)) * K + k0 + col];
        const unsigned int loff =
            (unsigned int)(uintptr_t)(ldsbuf + row * LDSROW + col);
        async_ld_b128(loff, g);
    }
}

// C = epilogue(A (MxK, row-major bf16) x B^T (B is NxK, row-major bf16))
template <int EPI>
__global__ __launch_bounds__(256) void gemm_bt_kernel(
    const __bf16* __restrict__ A, const __bf16* __restrict__ B,
    void* __restrict__ Cout, int M, int N, int K,
    long long strideAb, long long strideBb, long long strideCb,
    const float* __restrict__ ep0,     // gamma / sparse_w
    const int* __restrict__ ep1_i,     // mask (B,L) int32
    const float* __restrict__ ep1_f)   // beta
{
    __shared__ __align__(16) __bf16 lds[2][NROWS * LDSROW]; // 60 KB

    const int b = blockIdx.z;
    A += (long long)b * strideAb;
    B += (long long)b * strideBb;

    const int bm = blockIdx.y * BM;
    const int bn = blockIdx.x * BN;
    const int t = threadIdx.x;
    const int lane = t & 31;
    const int wid = t >> 5;
    const int wm = (wid & 1) * 64;    // wave M offset within tile (2 waves)
    const int wn = (wid >> 1) * 64;   // wave N offset within tile (4 waves)
    const int half = lane >> 4;
    const int r16 = lane & 15;

    v8f acc[4][4];
    const v8f vzero = {0.f, 0.f, 0.f, 0.f, 0.f, 0.f, 0.f, 0.f};
#pragma unroll
    for (int i = 0; i < 4; i++)
#pragma unroll
        for (int j = 0; j < 4; j++) acc[i][j] = vzero;

    // Prologue: DMA tile 0 into buffer 0.
    issue_tile_async(A, B, bm, bn, K, 0, &lds[0][0], t);

    int cur = 0;
    for (int k0 = 0; k0 < K; k0 += BK) {
        wait_async0();        // this wave's DMA done
        __syncthreads();      // everyone's DMA done; prev reads of lds[cur^1] done
        if (k0 + BK < K)      // overlap next tile's DMA with this tile's WMMAs
            issue_tile_async(A, B, bm, bn, K, k0 + BK, &lds[cur ^ 1][0], t);

        const __bf16* buf = &lds[cur][0];
        v16bf af[4], bfr[4];
#pragma unroll
        for (int i = 0; i < 4; i++)
            af[i] = load_frag_a(buf, wm + i * 16 + r16, half);
#pragma unroll
        for (int j = 0; j < 4; j++)
            bfr[j] = load_frag_b(buf, BM + wn + j * 16 + r16, half);

#pragma unroll
        for (int i = 0; i < 4; i++)
#pragma unroll
            for (int j = 0; j < 4; j++)
                acc[i][j] = __builtin_amdgcn_wmma_f32_16x16x32_bf16(
                    false, af[i], false, bfr[j], (short)0, acc[i][j], false, false);
        cur ^= 1;
    }

    // Epilogue. C fragment: VGPR e, lanes 0-15 -> M=e, lanes 16-31 -> M=e+8; N=lane&15.
#pragma unroll
    for (int i = 0; i < 4; i++) {
#pragma unroll
        for (int j = 0; j < 4; j++) {
#pragma unroll
            for (int e = 0; e < 8; e++) {
                const int m = bm + wm + i * 16 + e + half * 8;
                const int n = bn + wn + j * 16 + r16;
                float val = acc[i][j][e];
                if constexpr (EPI == EPI_SILU) {
                    __bf16* C = (__bf16*)Cout;
                    C[(long long)m * N + n] = (__bf16)silu_f(val);
                } else if constexpr (EPI == EPI_SCALE) {
                    __bf16* C = (__bf16*)Cout;
                    val = val * ep0[n] + ep1_f[n];
                    C[(long long)m * N + n] = (__bf16)val;
                } else if constexpr (EPI == EPI_ATTN) {
                    __bf16* C = (__bf16*)Cout + (long long)b * strideCb;
                    // masked (not allowed) = (m!=n) && (pad || future)
                    const bool masked =
                        (m != n) && ((ep1_i[(long long)b * LSEQ + n] == 0) || (n > m));
                    float v = masked ? 0.f : val;
                    v *= ep0[(long long)m * LSEQ + n];
                    v = fmaxf(v, 0.f);
                    v = v * v * (1.0f / (1024.0f * 1024.0f));
                    C[(long long)m * N + n] = (__bf16)v;
                } else if constexpr (EPI == EPI_SILU_T) {
                    // store v transposed: v_t[b2][n][l] , m = b2*LSEQ + l
                    __bf16* C = (__bf16*)Cout;
                    const int b2 = m >> 10;
                    const int l = m & (LSEQ - 1);
                    C[((long long)b2 * N + n) * LSEQ + l] = (__bf16)silu_f(val);
                } else { // EPI_OUT
                    float* C = (float*)Cout + (long long)b * strideCb;
                    C[(long long)m * N + n] = val;
                }
            }
        }
    }
}

// x[b,l,:] = bf16(item_emb[positives[b,l]] + pos_emb[l]); one block per row
__global__ __launch_bounds__(256) void gather_x_kernel(
    const int* __restrict__ positives, const float* __restrict__ item_emb,
    const float* __restrict__ pos_emb, __bf16* __restrict__ xbf)
{
    const int row = blockIdx.x;           // 0 .. B*L-1
    const int l = row & (LSEQ - 1);
    const long long item = positives[row];
    const int c = threadIdx.x * 4;
    const float4 ie = *(const float4*)&item_emb[item * 1024 + c];
    const float4 pe = *(const float4*)&pos_emb[(long long)l * 1024 + c];
    v4bf o;
    o[0] = (__bf16)(ie.x + pe.x);
    o[1] = (__bf16)(ie.y + pe.y);
    o[2] = (__bf16)(ie.z + pe.z);
    o[3] = (__bf16)(ie.w + pe.w);
    *(v4bf*)&xbf[(long long)row * 1024 + c] = o;
}

__global__ __launch_bounds__(256) void f32_to_bf16_kernel(
    const float* __restrict__ in, __bf16* __restrict__ out, int n)
{
    const int i = (blockIdx.x * 256 + threadIdx.x) * 4;
    if (i < n) {
        const float4 f = *(const float4*)&in[i];
        v4bf o;
        o[0] = (__bf16)f.x; o[1] = (__bf16)f.y; o[2] = (__bf16)f.z; o[3] = (__bf16)f.w;
        *(v4bf*)&out[i] = o;
    }
}

extern "C" void kernel_launch(void* const* d_in, const int* in_sizes, int n_in,
                              void* d_out, int out_size, void* d_ws, size_t ws_size,
                              hipStream_t stream) {
    (void)in_sizes; (void)n_in; (void)out_size; (void)ws_size;
    const int*   positives = (const int*)  d_in[0];
    const int*   mask      = (const int*)  d_in[1];
    const float* item_emb  = (const float*)d_in[2];
    const float* pos_emb   = (const float*)d_in[3];
    const float* Wz        = (const float*)d_in[4];
    const float* Wv        = (const float*)d_in[5];
    const float* Wq        = (const float*)d_in[6];
    const float* Wk        = (const float*)d_in[7];
    const float* gamma_q   = (const float*)d_in[8];   // (HEADS,L): head 0 = first L
    const float* beta_q    = (const float*)d_in[9];
    const float* gamma_k   = (const float*)d_in[10];
    const float* beta_k    = (const float*)d_in[11];
    const float* sparse_w  = (const float*)d_in[12];

    const int Bsz = 32, L = 1024, D = 1024;
    const long long BL = (long long)Bsz * L;         // 32768
    const long long LL = (long long)L * L;           // 1M

    // Workspace carve-out (bf16): x(64M) + 4 weights(8M) + z/a(64M) + q/vT(64M) + k(64M) = 264MB
    char* ws = (char*)d_ws;
    size_t off = 0;
    __bf16* xbf = (__bf16*)(ws + off); off += (size_t)BL * D * 2;
    __bf16* wzb = (__bf16*)(ws + off); off += (size_t)LL * 2;
    __bf16* wvb = (__bf16*)(ws + off); off += (size_t)LL * 2;
    __bf16* wqb = (__bf16*)(ws + off); off += (size_t)LL * 2;
    __bf16* wkb = (__bf16*)(ws + off); off += (size_t)LL * 2;
    __bf16* zbf = (__bf16*)(ws + off); off += (size_t)Bsz * LL * 2; // later reused as 'a'
    __bf16* qbf = (__bf16*)(ws + off); off += (size_t)Bsz * LL * 2; // later reused as v^T
    __bf16* kbf = (__bf16*)(ws + off); off += (size_t)Bsz * LL * 2;

    // weights -> bf16
    f32_to_bf16_kernel<<<1024, 256, 0, stream>>>(Wz, wzb, (int)LL);
    f32_to_bf16_kernel<<<1024, 256, 0, stream>>>(Wv, wvb, (int)LL);
    f32_to_bf16_kernel<<<1024, 256, 0, stream>>>(Wq, wqb, (int)LL);
    f32_to_bf16_kernel<<<1024, 256, 0, stream>>>(Wk, wkb, (int)LL);

    // x = item_emb[positives] + pos_emb
    gather_x_kernel<<<(int)BL, 256, 0, stream>>>(positives, item_emb, pos_emb, xbf);

    const dim3 gBig(D / BN, (unsigned)(BL / BM), 1);  // (4, 256, 1)
    const dim3 gBat(L / BN, L / BM, Bsz);             // (4, 8, 32)

    // z = silu(x @ Wz^T)
    gemm_bt_kernel<EPI_SILU><<<gBig, 256, 0, stream>>>(
        xbf, wzb, zbf, (int)BL, L, D, 0, 0, 0, nullptr, nullptr, nullptr);
    // q = (z @ Wq^T) * gamma_q0 + beta_q0
    gemm_bt_kernel<EPI_SCALE><<<gBig, 256, 0, stream>>>(
        zbf, wqb, qbf, (int)BL, L, L, 0, 0, 0, gamma_q, nullptr, beta_q);
    // k = (z @ Wk^T) * gamma_k0 + beta_k0
    gemm_bt_kernel<EPI_SCALE><<<gBig, 256, 0, stream>>>(
        zbf, wkb, kbf, (int)BL, L, L, 0, 0, 0, gamma_k, nullptr, beta_k);
    // a = relu((q @ k^T) masked * sparse_w)^2 / (L*D)   (overwrites z)
    gemm_bt_kernel<EPI_ATTN><<<gBat, 256, 0, stream>>>(
        qbf, kbf, zbf, L, L, L, LL, LL, LL, sparse_w, mask, nullptr);
    // v^T = silu(x @ Wv^T)^T   (overwrites q)
    gemm_bt_kernel<EPI_SILU_T><<<gBig, 256, 0, stream>>>(
        xbf, wvb, qbf, (int)BL, D, D, 0, 0, 0, nullptr, nullptr, nullptr);
    // out = a @ v  ==  a @ (v^T)^T   (fp32 output)
    gemm_bt_kernel<EPI_OUT><<<gBat, 256, 0, stream>>>(
        zbf, qbf, d_out, L, D, L, LL, (long long)D * L, (long long)L * D,
        nullptr, nullptr, nullptr);
}
